// FlashMoEModel_44641890075032
// MI455X (gfx1250) — matrix operand order repaired and checked by reading, not compiled, not run on hardware
//
#include <hip/hip_runtime.h>
#include <stdint.h>

#define D_MODEL   512
#define NUM_EXP   16
#define D_HIDDEN  1024
#define B_TOK     32768
#define CAPACITY  5120            // ceil(1.25 * 2*32768 / 16)
#define NASSIGN   (B_TOK * 2)

// ---- CDNA5 async global->LDS path (device compile only) ----
#if defined(__HIP_DEVICE_COMPILE__) && __has_builtin(__builtin_amdgcn_global_load_async_to_lds_b128)
#define USE_ASYNC 1
#else
#define USE_ASYNC 0
#endif

#if USE_ASYNC
#define NBUF 4                           // power of two -> cheap (k & 3) indexing
#else
#define NBUF 2
#endif

// LDS tile geometry: A tile 128x32 bf16, B tile 256x32 bf16, padded row stride
#define AROWS 128
#define BROWS 256
#define LSTRIDE 40                       // ushorts per row (80B, conflict-free)
#define ABUF_USH (AROWS * LSTRIDE)       // 5120 ushorts
#define BBUF_USH (BROWS * LSTRIDE)       // 10240 ushorts
#define BUF_USH  (ABUF_USH + BBUF_USH)   // 15360 ushorts / buffer

typedef __attribute__((ext_vector_type(16))) __bf16 v16bf;
typedef __attribute__((ext_vector_type(8)))  float  v8f;
typedef __attribute__((__vector_size__(16))) int    v4i;

union FragAB { uint4 u[2]; v16bf v; };

static __device__ __forceinline__ unsigned short f32_to_bf16_rne(float f) {
    union { float f; unsigned int u; } c; c.f = f;
    unsigned int u = c.u;
    unsigned int lsb = (u >> 16) & 1u;
    u += 0x7fffu + lsb;
    return (unsigned short)(u >> 16);
}

#if USE_ASYNC
// builtin signature: (v4i AS1* src, v4i AS3* dst, Ii offset, Ii cpol).
// Generic int4* args; clang's builtin rewriting + InferAddressSpaces resolve AS.
static __device__ __forceinline__ void async_b128(const void* g, void* l) {
    __builtin_amdgcn_global_load_async_to_lds_b128((v4i*)g, (v4i*)l, 0, 0);
}
static __device__ __forceinline__ void wait_async_le6() {
#if __has_builtin(__builtin_amdgcn_s_wait_asynccnt)
    __builtin_amdgcn_s_wait_asynccnt(6);
#else
    asm volatile("s_wait_asynccnt 0x6" ::: "memory");
#endif
}
static __device__ __forceinline__ void wait_async_le0() {
#if __has_builtin(__builtin_amdgcn_s_wait_asynccnt)
    __builtin_amdgcn_s_wait_asynccnt(0);
#else
    asm volatile("s_wait_asynccnt 0x0" ::: "memory");
#endif
}
#endif

// ---------------- casts ----------------
__global__ void cast_f32_bf16(const float* __restrict__ src,
                              unsigned short* __restrict__ dst, int n) {
    int i = blockIdx.x * blockDim.x + threadIdx.x;
    int stride = gridDim.x * blockDim.x;
    for (; i < n; i += stride) dst[i] = f32_to_bf16_rne(src[i]);
}

__global__ void zero_counts(int* counts) {
    if (threadIdx.x < NUM_EXP) counts[threadIdx.x] = 0;
}

// ---------------- tiled bf16 WMMA GEMM ----------------
// C[M,N] = A[M,K] * Bw[N,K]^T (+bias)(relu?), batched over blockIdx.z.
// Block tile 128x256, 8 waves as 2(M)x4(N), wave tile 64x64 = 4x4 wmma tiles.
// K-step 32. Quad-buffered LDS with async copies (fallback: reg staging).
template<bool RELU, bool OUT_F32, bool OUT_BF16, bool HAS_BIAS>
__global__ __launch_bounds__(256) void wmma_gemm_bf16(
    const unsigned short* __restrict__ Abase,
    const unsigned short* __restrict__ Bbase,
    const float* __restrict__ bias,
    float* __restrict__ Cf,
    unsigned short* __restrict__ Ch,
    int N, int K,
    long strideA, long strideB, long strideCf, long strideCh,
    const int* __restrict__ rowLimit)
{
    const int e  = blockIdx.z;
    const int m0 = blockIdx.y * 128;
    const int n0 = blockIdx.x * 256;
    if (rowLimit && m0 >= rowLimit[e]) return;   // dead row tile for this expert

    const unsigned short* A  = Abase + (long)e * strideA;
    const unsigned short* Bw = Bbase + (long)e * strideB;

    __shared__ unsigned short smem[NBUF * BUF_USH];

    const int tid  = threadIdx.x;
    const int lane = tid & 31;
    const int wave = tid >> 5;
    const int wm   = wave & 1;   // 2 waves along M
    const int wn   = wave >> 1;  // 4 waves along N
    const int half = lane >> 4;
    const int r16  = lane & 15;

    v8f acc[4][4];
    #pragma unroll
    for (int mi = 0; mi < 4; ++mi)
        #pragma unroll
        for (int ni = 0; ni < 4; ++ni)
            #pragma unroll
            for (int g = 0; g < 8; ++g) acc[mi][ni][g] = 0.0f;

    // staging map: thread stages one 32B half-row of A and two of B
    const int srow  = tid >> 1;          // 0..127
    const int shalf = tid & 1;
    const unsigned short* gA0 = A  + (long)(m0 + srow) * K + shalf * 16;
    const unsigned short* gB0 = Bw + (long)(n0 + srow) * K + shalf * 16;
    const unsigned short* gB1 = Bw + (long)(n0 + 128 + srow) * K + shalf * 16;
    const int lAoff = srow * LSTRIDE + shalf * 16;
    const int lBoff = ABUF_USH + srow * LSTRIDE + shalf * 16;

    const int nsteps = K >> 5;

#if USE_ASYNC
    auto issue = [&](int kk, int b) {
        unsigned short* base = smem + b * BUF_USH;
        const int k0 = kk << 5;
        async_b128(gA0 + k0,      base + lAoff);
        async_b128(gA0 + k0 + 8,  base + lAoff + 8);
        async_b128(gB0 + k0,      base + lBoff);
        async_b128(gB0 + k0 + 8,  base + lBoff + 8);
        async_b128(gB1 + k0,      base + lBoff + 128 * LSTRIDE);
        async_b128(gB1 + k0 + 8,  base + lBoff + 128 * LSTRIDE + 8);
    };
#else
    uint4 st[6];
    auto loadregs = [&](int kk) {
        const int k0 = kk << 5;
        st[0] = *(const uint4*)(gA0 + k0);  st[1] = *(const uint4*)(gA0 + k0 + 8);
        st[2] = *(const uint4*)(gB0 + k0);  st[3] = *(const uint4*)(gB0 + k0 + 8);
        st[4] = *(const uint4*)(gB1 + k0);  st[5] = *(const uint4*)(gB1 + k0 + 8);
    };
    auto storeLds = [&](int b) {
        unsigned short* base = smem + b * BUF_USH;
        *(uint4*)(base + lAoff)     = st[0];  *(uint4*)(base + lAoff + 8) = st[1];
        *(uint4*)(base + lBoff)     = st[2];  *(uint4*)(base + lBoff + 8) = st[3];
        *(uint4*)(base + lBoff + 128 * LSTRIDE)     = st[4];
        *(uint4*)(base + lBoff + 128 * LSTRIDE + 8) = st[5];
    };
#endif

    auto compute = [&](int b) {
        const unsigned short* bA = smem + b * BUF_USH;
        const unsigned short* bB = bA + ABUF_USH;
        FragAB af[4];
        #pragma unroll
        for (int mi = 0; mi < 4; ++mi) {
            const unsigned short* base = bA + (wm * 64 + mi * 16 + r16) * LSTRIDE;
            af[mi].u[0] = *(const uint4*)(base + half * 8);
            af[mi].u[1] = *(const uint4*)(base + 16 + half * 8);
        }
        FragAB bfr[4];
        #pragma unroll
        for (int ni = 0; ni < 4; ++ni) {
            const unsigned short* base = bB + (wn * 64 + ni * 16 + r16) * LSTRIDE;
            bfr[ni].u[0] = *(const uint4*)(base + half * 8);
            bfr[ni].u[1] = *(const uint4*)(base + 16 + half * 8);
        }
        #pragma unroll
        for (int mi = 0; mi < 4; ++mi)
            #pragma unroll
            for (int ni = 0; ni < 4; ++ni)
                acc[mi][ni] = __builtin_amdgcn_wmma_f32_16x16x32_bf16(
                    false, af[mi].v, false, bfr[ni].v,
                    (short)0, acc[mi][ni], false, false);
    };

#if USE_ASYNC
    // quad buffer, prefetch depth 2, one barrier per K-step
    issue(0, 0);
    if (nsteps > 1) issue(1, 1);
    for (int k = 0; k < nsteps; ++k) {
        if (k + 1 < nsteps) wait_async_le6(); else wait_async_le0();
        __syncthreads();                  // dscnt0 fence + barrier
        compute(k & 3);
        if (k + 2 < nsteps) issue(k + 2, (k + 2) & 3);
    }
#else
    // double buffer with register staging, one barrier per K-step
    loadregs(0);
    storeLds(0);
    if (nsteps > 1) loadregs(1);
    for (int k = 0; k < nsteps; ++k) {
        __syncthreads();                  // dscnt0 fence + barrier
        compute(k & 1);
        if (k + 1 < nsteps) {
            storeLds((k + 1) & 1);
            if (k + 2 < nsteps) loadregs(k + 2);
        }
    }
#endif

    // epilogue: VGPR g holds row (half*8+g), lane r16 is column
    #pragma unroll
    for (int mi = 0; mi < 4; ++mi) {
        #pragma unroll
        for (int ni = 0; ni < 4; ++ni) {
            const int col = n0 + wn * 64 + ni * 16 + r16;
            float bv = 0.0f;
            if (HAS_BIAS) bv = bias[col];
            #pragma unroll
            for (int g = 0; g < 8; ++g) {
                const int row = m0 + wm * 64 + mi * 16 + half * 8 + g;
                float v = acc[mi][ni][g] + bv;
                if (RELU) v = fmaxf(v, 0.0f);
                if (OUT_F32)  Cf[(long)e * strideCf + (long)row * N + col] = v;
                if (OUT_BF16) Ch[(long)e * strideCh + (long)row * N + col] = f32_to_bf16_rne(v);
            }
        }
    }
}

// ---------------- router: one wave per token ----------------
__global__ __launch_bounds__(256) void router_topk(
    const float* __restrict__ enc, const float* __restrict__ Wg,
    const float* __restrict__ bg,
    int* __restrict__ counts, int* __restrict__ aExp,
    int* __restrict__ aSlot, float* __restrict__ aW)
{
    __shared__ float sWg[NUM_EXP * D_MODEL];   // 32 KB
    for (int i = threadIdx.x; i < NUM_EXP * D_MODEL; i += blockDim.x) sWg[i] = Wg[i];
    __syncthreads();

    const int wave = threadIdx.x >> 5, lane = threadIdx.x & 31;
    const int t = blockIdx.x * 8 + wave;
    if (t >= B_TOK) return;

    float acc[NUM_EXP];
    #pragma unroll
    for (int k = 0; k < NUM_EXP; ++k) acc[k] = 0.0f;

    const float* xr = enc + (long)t * D_MODEL;
    for (int d = lane; d < D_MODEL; d += 32) {
        float xv = xr[d];
        #pragma unroll
        for (int k = 0; k < NUM_EXP; ++k) acc[k] += xv * sWg[k * D_MODEL + d];
    }
    #pragma unroll
    for (int k = 0; k < NUM_EXP; ++k) {
        float v = acc[k];
        #pragma unroll
        for (int off = 16; off > 0; off >>= 1) v += __shfl_xor(v, off, 32);
        acc[k] = v + bg[k];
    }
    if (lane == 0) {
        float m1 = -1e30f, m2 = -1e30f; int i1 = 0, i2 = 0;
        #pragma unroll
        for (int k = 0; k < NUM_EXP; ++k) {
            float v = acc[k];
            if (v > m1) { m2 = m1; i2 = i1; m1 = v; i1 = k; }
            else if (v > m2) { m2 = v; i2 = k; }
        }
        float e1 = expf(m2 - m1);                 // stable softmax, tau=1
        float denom = 1.0f + e1 + 1e-12f;
        float w0 = 1.0f / denom, w1 = e1 / denom;
        int s0 = atomicAdd(&counts[i1], 1);
        int s1 = atomicAdd(&counts[i2], 1);
        aExp[2 * t] = i1;     aSlot[2 * t] = s0;     aW[2 * t] = w0;
        aExp[2 * t + 1] = i2; aSlot[2 * t + 1] = s1; aW[2 * t + 1] = w1;
    }
}

// ---------------- dispatch: copy encoded row into expert buffer ----------------
__global__ __launch_bounds__(64) void dispatch_copy(
    const unsigned short* __restrict__ encH,
    const int* __restrict__ aExp, const int* __restrict__ aSlot,
    const float* __restrict__ aW,
    unsigned short* __restrict__ buf)
{
    const int aid = blockIdx.x;
    const int s = aSlot[aid];
    const float w = aW[aid];
    if (s >= CAPACITY || !(w > 0.0f)) return;
    const int e = aExp[aid];
    const int t = aid >> 1;
    const uint4* src = (const uint4*)(encH + (long)t * D_MODEL);
    uint4* dst = (uint4*)(buf + ((long)e * CAPACITY + s) * D_MODEL);
    dst[threadIdx.x] = src[threadIdx.x];   // 64 threads x 16B = 1KB row
}

// ---------------- deterministic combine ----------------
__global__ __launch_bounds__(128) void combine_out(
    const float* __restrict__ enc, const float* __restrict__ delta,
    const int* __restrict__ aExp, const int* __restrict__ aSlot,
    const float* __restrict__ aW,
    float* __restrict__ out)
{
    const int t = blockIdx.x;
    const int d = threadIdx.x * 4;
    float4 v = *(const float4*)(enc + (long)t * D_MODEL + d);
    #pragma unroll
    for (int k = 0; k < 2; ++k) {
        const int aid = 2 * t + k;
        const int s = aSlot[aid];
        const float w = aW[aid];
        if (s < CAPACITY && w > 0.0f) {
            const int e = aExp[aid];
            const float4 dv = *(const float4*)(delta + ((long)e * CAPACITY + s) * D_MODEL + d);
            v.x += w * dv.x; v.y += w * dv.y; v.z += w * dv.z; v.w += w * dv.w;
        }
    }
    *(float4*)(out + (long)t * D_MODEL + d) = v;
}

// ---------------- host ----------------
extern "C" void kernel_launch(void* const* d_in, const int* in_sizes, int n_in,
                              void* d_out, int out_size, void* d_ws, size_t ws_size,
                              hipStream_t stream)
{
    (void)in_sizes; (void)n_in; (void)out_size; (void)ws_size;
    const float* x  = (const float*)d_in[0];
    const float* We = (const float*)d_in[1];
    const float* be = (const float*)d_in[2];
    const float* Wg = (const float*)d_in[3];
    const float* bg = (const float*)d_in[4];
    const float* W1 = (const float*)d_in[5];
    const float* W2 = (const float*)d_in[6];
    float* y = (float*)d_out;

    char* ws = (char*)d_ws;
    size_t off = 0;
    auto alloc = [&](size_t bytes) -> char* {
        char* p = ws + off;
        off = (off + bytes + 255) & ~(size_t)255;
        return p;
    };

    unsigned short* xh   = (unsigned short*)alloc((size_t)B_TOK * D_MODEL * 2);
    unsigned short* WeH  = (unsigned short*)alloc((size_t)D_MODEL * D_MODEL * 2);
    unsigned short* W1H  = (unsigned short*)alloc((size_t)NUM_EXP * D_HIDDEN * D_MODEL * 2);
    unsigned short* W2H  = (unsigned short*)alloc((size_t)NUM_EXP * D_MODEL * D_HIDDEN * 2);
    float*          encF = (float*)alloc((size_t)B_TOK * D_MODEL * 4);
    unsigned short* encH = (unsigned short*)alloc((size_t)B_TOK * D_MODEL * 2);
    unsigned short* buf  = (unsigned short*)alloc((size_t)NUM_EXP * CAPACITY * D_MODEL * 2);
    unsigned short* hid  = (unsigned short*)alloc((size_t)NUM_EXP * CAPACITY * D_HIDDEN * 2);
    float*          dlt  = (float*)alloc((size_t)NUM_EXP * CAPACITY * D_MODEL * 4);
    int*            cnt  = (int*)alloc(NUM_EXP * 4);
    int*            aExp = (int*)alloc((size_t)NASSIGN * 4);
    int*            aSlot= (int*)alloc((size_t)NASSIGN * 4);
    float*          aW   = (float*)alloc((size_t)NASSIGN * 4);

    zero_counts<<<1, 32, 0, stream>>>(cnt);
    cast_f32_bf16<<<2048, 256, 0, stream>>>(x,  xh,  B_TOK * D_MODEL);
    cast_f32_bf16<<<512,  256, 0, stream>>>(We, WeH, D_MODEL * D_MODEL);
    cast_f32_bf16<<<2048, 256, 0, stream>>>(W1, W1H, NUM_EXP * D_HIDDEN * D_MODEL);
    cast_f32_bf16<<<2048, 256, 0, stream>>>(W2, W2H, NUM_EXP * D_MODEL * D_HIDDEN);

    // 1. encoder: enc = x @ We^T + be  (f32 + bf16 copies)
    wmma_gemm_bf16<false, true, true, true>
        <<<dim3(D_MODEL / 256, B_TOK / 128, 1), 256, 0, stream>>>(
            xh, WeH, be, encF, encH, D_MODEL, D_MODEL,
            0, 0, 0, 0, nullptr);

    // 2. router top-2 + masked softmax + slot append
    router_topk<<<B_TOK / 8, 256, 0, stream>>>(encF, Wg, bg, cnt, aExp, aSlot, aW);

    // 3. dispatch rows into per-expert buffers
    dispatch_copy<<<NASSIGN, 64, 0, stream>>>(encH, aExp, aSlot, aW, buf);

    // 4. expert layer 1: hid = relu(buf @ W1^T) -> bf16
    wmma_gemm_bf16<true, false, true, false>
        <<<dim3(D_HIDDEN / 256, CAPACITY / 128, NUM_EXP), 256, 0, stream>>>(
            buf, W1H, nullptr, nullptr, hid, D_HIDDEN, D_MODEL,
            (long)CAPACITY * D_MODEL, (long)D_HIDDEN * D_MODEL,
            0, (long)CAPACITY * D_HIDDEN, cnt);

    // 5. expert layer 2: delta = hid @ W2^T -> f32
    wmma_gemm_bf16<false, true, false, false>
        <<<dim3(D_MODEL / 256, CAPACITY / 128, NUM_EXP), 256, 0, stream>>>(
            hid, W2H, nullptr, dlt, nullptr, D_MODEL, D_HIDDEN,
            (long)CAPACITY * D_HIDDEN, (long)D_MODEL * D_HIDDEN,
            (long)CAPACITY * D_MODEL, 0, cnt);

    // 6. combine: y = enc + w0*delta0 + w1*delta1 (fixed order -> deterministic)
    combine_out<<<B_TOK, 128, 0, stream>>>(encF, dlt, aExp, aSlot, aW, y);
}
